// MultiHeadAttention_12051678233275
// MI455X (gfx1250) — matrix operand order repaired and checked by reading, not compiled, not run on hardware
//
#include <hip/hip_runtime.h>
#include <hip/hip_bf16.h>
#include <stdint.h>

// ---------------------------------------------------------------------------
// MI455X / gfx1250 fused MHA: bf16 WMMA path (v_wmma_f32_16x16x32_bf16)
// B=4, S=2048, D=1024, H=16, DH=64
// All WMMA operand fragments are loaded as contiguous 16B chunks from LDS
// (A tiles row-major [m][k], B tiles N-major [n][k]) so they lower to
// ds_load_b128 instead of strided ds_load_u16 gathers.
// ---------------------------------------------------------------------------

#define BATCH 4
#define SEQ   2048
#define DIM   1024
#define HEADS 16
#define DH    64
#define BH    (BATCH * HEADS)

typedef __bf16 v16bf __attribute__((ext_vector_type(16)));
typedef float  v8f   __attribute__((ext_vector_type(8)));

static __device__ __forceinline__ v8f vzero() {
    v8f z = {0.f, 0.f, 0.f, 0.f, 0.f, 0.f, 0.f, 0.f};
    return z;
}

static __device__ __forceinline__ v8f wmma_bf16(v16bf a, v16bf b, v8f c) {
    // 8-arg form: (neg_a, A, neg_b, B, c_mod, C, reuse_a, reuse_b)
    return __builtin_amdgcn_wmma_f32_16x16x32_bf16(false, a, false, b, (short)0, c, false, false);
}

// A-matrix fragment, 16x32 bf16, row-major source [m][k] with leading dim `ld`.
// lanes 0-15 -> M=lane, K=0..15 ; lanes 16-31 -> M=lane-16, K=16..31 (matches load_bn_frag)
// 16 contiguous bf16 (32B) per lane -> 2x ds_load_b128.
static __device__ __forceinline__ v16bf load_a_frag(const __bf16* __restrict__ base, int ld, int lane) {
    const int m = lane & 15;
    const int khalf = (lane >> 4) << 4;
    v16bf f;
#pragma unroll
    for (int i = 0; i < 8; ++i) {
        const int kk = khalf + i * 2;
        f[2 * i]     = base[m * ld + kk];
        f[2 * i + 1] = base[m * ld + kk + 1];
    }
    return f;
}

// B-matrix fragment, 32x16 bf16, from an N-major tile: src[n][k], leading dim ld.
// lanes 0-15 -> N=lane, K=0..15 ; lanes 16-31 -> N=lane-16, K=16..31
// 16 contiguous bf16 (32B) per lane -> 2x ds_load_b128.
static __device__ __forceinline__ v16bf load_bn_frag(const __bf16* __restrict__ base, int ld, int lane) {
    const int n = lane & 15;
    const int kb = (lane >> 4) << 4;
    v16bf f;
#pragma unroll
    for (int j = 0; j < 16; ++j) f[j] = base[n * ld + kb + j];
    return f;
}

// ---------------------------------------------------------------------------
// Kernel 1: fused  Z = relu((X @ W + b)[head] @ Wf + bf) * scale  -> bf16 [BH][S][DH]
// grid = (HEADS, (B*S)/128), block = 256 (8 waves, 16 rows each)
// ---------------------------------------------------------------------------
__global__ __launch_bounds__(256) void proj_ff_kernel(
    const float* __restrict__ X, const float* __restrict__ W, const float* __restrict__ bias,
    const float* __restrict__ Wf, const float* __restrict__ bf,
    __bf16* __restrict__ Out, float scale)
{
    __shared__ __align__(16) __bf16 Asl[128][40];    // A tile  [m][k]
    __shared__ __align__(16) __bf16 Bsl[64][40];     // W tile  [n][k]  (transposed at staging)
    __shared__ __align__(16) __bf16 Ysl[128][72];    // Y tile  [m][k]
    __shared__ __align__(16) __bf16 Wfs[64][72];     // Wf      [n][k]  (transposed at staging)

    const int t    = threadIdx.x;
    const int lane = t & 31;
    const int w    = t >> 5;
    const int h    = blockIdx.x;
    const int m0   = blockIdx.y * 128;
    const int nc0  = h * DH;

    // stage Wf once, N-major: Wfs[n][k] = Wf[k][n]
#pragma unroll
    for (int i = 0; i < 16; ++i) {
        const int l = i * 256 + t;                 // 4096 elements
        const int kk = l >> 6, nn = l & 63;        // global read coalesced over nn
        Wfs[nn][kk] = (__bf16)Wf[kk * DH + nn];
    }

    v8f acc[4];
#pragma unroll
    for (int nt = 0; nt < 4; ++nt) acc[nt] = vzero();

    for (int k0 = 0; k0 < DIM; k0 += 32) {
        __syncthreads();
        // A tile: 128x32 f32 -> bf16, 4 contiguous cols per thread
#pragma unroll
        for (int i = 0; i < 4; ++i) {
            const int lq = i * 256 + t;            // 1024 chunks of 4
            const int r = lq >> 3, c4 = (lq & 7) * 4;
            const float4 xv = *(const float4*)(X + (size_t)(m0 + r) * DIM + k0 + c4);
            Asl[r][c4 + 0] = (__bf16)xv.x;
            Asl[r][c4 + 1] = (__bf16)xv.y;
            Asl[r][c4 + 2] = (__bf16)xv.z;
            Asl[r][c4 + 3] = (__bf16)xv.w;
        }
        // W tile: 32x64, stored N-major Bsl[n][k]
#pragma unroll
        for (int i = 0; i < 8; ++i) {
            const int l = i * 256 + t;             // 2048 elements
            const int r = l >> 6, c = l & 63;      // r = k, c = n (coalesced read)
            Bsl[c][r] = (__bf16)W[(size_t)(k0 + r) * DIM + nc0 + c];
        }
        __syncthreads();

        const v16bf a = load_a_frag(&Asl[w * 16][0], 40, lane);
#pragma unroll
        for (int nt = 0; nt < 4; ++nt) {
            const v16bf bm = load_bn_frag(&Bsl[nt * 16][0], 40, lane);
            acc[nt] = wmma_bf16(a, bm, acc[nt]);
        }
    }

    const int n  = lane & 15;
    const int mb = (lane >> 4) << 3;

    // add bias, stage Y (bf16) for the per-head FF GEMM
#pragma unroll
    for (int nt = 0; nt < 4; ++nt) {
        const float bv = bias[nc0 + nt * 16 + n];
#pragma unroll
        for (int i = 0; i < 8; ++i)
            Ysl[w * 16 + mb + i][nt * 16 + n] = (__bf16)(acc[nt][i] + bv);
    }
    asm volatile("s_wait_dscnt 0" ::: "memory");   // intra-wave LDS RAW

    const v16bf ya0 = load_a_frag(&Ysl[w * 16][0], 72, lane);
    const v16bf ya1 = load_a_frag(&Ysl[w * 16][32], 72, lane);

    v8f acc2[4];
#pragma unroll
    for (int nt = 0; nt < 4; ++nt) {
        acc2[nt] = vzero();
        const v16bf b0 = load_bn_frag(&Wfs[nt * 16][0], 72, lane);
        const v16bf b1 = load_bn_frag(&Wfs[nt * 16][32], 72, lane);
        acc2[nt] = wmma_bf16(ya0, b0, acc2[nt]);
        acc2[nt] = wmma_bf16(ya1, b1, acc2[nt]);
    }

    // bias + relu + scale, store bf16 head-major [BH][S][DH]
#pragma unroll
    for (int nt = 0; nt < 4; ++nt) {
        const float bfv = bf[nt * 16 + n];
#pragma unroll
        for (int i = 0; i < 8; ++i) {
            const int r  = m0 + w * 16 + mb + i;     // global token
            const int bb = r >> 11;                  // / SEQ
            const int s  = r & (SEQ - 1);
            float z = acc2[nt][i] + bfv;
            z = z > 0.f ? z : 0.f;
            z *= scale;
            Out[((size_t)(bb * HEADS + h) * SEQ + s) * DH + nt * 16 + n] = (__bf16)z;
        }
    }
}

// ---------------------------------------------------------------------------
// Kernel 2: flash attention + output projection.
// grid = (S/128, BH), block = 256. Q already pre-scaled by 1/sqrt(DH).
// ---------------------------------------------------------------------------
__global__ __launch_bounds__(256) void attn_kernel(
    const __bf16* __restrict__ Qh, const __bf16* __restrict__ Kh, const __bf16* __restrict__ Vh,
    const int* __restrict__ mask, const float* __restrict__ Wo, const float* __restrict__ bo,
    float* __restrict__ out)
{
    __shared__ __align__(16) __bf16 Kt[64][72];    // K tile [key][dh] == N-major for Q@K^T
    __shared__ __align__(16) __bf16 Vt[64][72];    // V tile [dh][key] (transposed at staging)
    __shared__ __align__(16) __bf16 Ps[128][72];   // P / O restage, row-major [m][k]
    __shared__ __align__(16) __bf16 Wos[64][72];   // Wo [n][k] (transposed at staging)

    const int t    = threadIdx.x;
    const int lane = t & 31;
    const int w    = t >> 5;
    const int bh   = blockIdx.y;
    const int q0   = blockIdx.x * 128;
    const int n    = lane & 15;
    const int mb   = (lane >> 4) << 3;

    // stage Wo once, N-major: Wos[n][k] = Wo[k][n]
#pragma unroll
    for (int i = 0; i < 16; ++i) {
        const int l = i * 256 + t;
        const int kk = l >> 6, nn = l & 63;
        Wos[nn][kk] = (__bf16)Wo[kk * DH + nn];
    }

    // Q fragments straight from global (row-major, ld = DH)
    const __bf16* qbase = Qh + ((size_t)bh * SEQ + q0 + w * 16) * DH;
    const v16bf qa0 = load_a_frag(qbase, DH, lane);
    const v16bf qa1 = load_a_frag(qbase + 32, DH, lane);

    float mstat[8], lstat[8];
    v8f oacc[4];
#pragma unroll
    for (int i = 0; i < 8; ++i) { mstat[i] = -3.0e38f; lstat[i] = 0.f; }
#pragma unroll
    for (int nt = 0; nt < 4; ++nt) oacc[nt] = vzero();

    for (int j = 0; j < SEQ; j += 64) {
        __syncthreads();
        const __bf16* kg = Kh + ((size_t)bh * SEQ + j) * DH;
        const __bf16* vg = Vh + ((size_t)bh * SEQ + j) * DH;
#pragma unroll
        for (int i = 0; i < 4; ++i) {
            const int l4 = i * 256 + t;              // 1024 chunks of 4 bf16
            const int r = l4 >> 4, c4 = (l4 & 15) * 4;
            // K tile: row-major copy, 8B vector
            *(uint2*)&Kt[r][c4] = *(const uint2*)(kg + (size_t)r * DH + c4);
            // V tile: transpose into [dh][key]; global read stays coalesced/vector
            const __bf16* vp = vg + (size_t)r * DH + c4;
            Vt[c4 + 0][r] = vp[0];
            Vt[c4 + 1][r] = vp[1];
            Vt[c4 + 2][r] = vp[2];
            Vt[c4 + 3][r] = vp[3];
        }
        __syncthreads();

        // scores S = Q @ K^T  (already scaled via Q)
        v8f sacc[4];
#pragma unroll
        for (int nt = 0; nt < 4; ++nt) {
            sacc[nt] = vzero();
            const v16bf b0 = load_bn_frag(&Kt[nt * 16][0], 72, lane);
            const v16bf b1 = load_bn_frag(&Kt[nt * 16][32], 72, lane);
            sacc[nt] = wmma_bf16(qa0, b0, sacc[nt]);
            sacc[nt] = wmma_bf16(qa1, b1, sacc[nt]);
        }

        // key-padding mask (column == key index)
#pragma unroll
        for (int nt = 0; nt < 4; ++nt) {
            const int mv = mask[(size_t)bh * SEQ + j + nt * 16 + n];
            if (mv == 0) {
#pragma unroll
                for (int i = 0; i < 8; ++i) sacc[nt][i] = -1e9f;
            }
        }

        // online softmax: row max over 16 columns (lanes within each half)
        float rm[8];
#pragma unroll
        for (int i = 0; i < 8; ++i)
            rm[i] = fmaxf(fmaxf(sacc[0][i], sacc[1][i]), fmaxf(sacc[2][i], sacc[3][i]));
#pragma unroll
        for (int d = 1; d < 16; d <<= 1)
#pragma unroll
            for (int i = 0; i < 8; ++i) rm[i] = fmaxf(rm[i], __shfl_xor(rm[i], d, 32));

        float mn[8], alpha[8];
#pragma unroll
        for (int i = 0; i < 8; ++i) {
            mn[i]    = fmaxf(mstat[i], rm[i]);
            alpha[i] = __expf(mstat[i] - mn[i]);
            mstat[i] = mn[i];
        }

        // P = exp(S - m), row sums
#pragma unroll
        for (int nt = 0; nt < 4; ++nt)
#pragma unroll
            for (int i = 0; i < 8; ++i) sacc[nt][i] = __expf(sacc[nt][i] - mn[i]);

        float rs[8];
#pragma unroll
        for (int i = 0; i < 8; ++i)
            rs[i] = sacc[0][i] + sacc[1][i] + sacc[2][i] + sacc[3][i];
#pragma unroll
        for (int d = 1; d < 16; d <<= 1)
#pragma unroll
            for (int i = 0; i < 8; ++i) rs[i] += __shfl_xor(rs[i], d, 32);
#pragma unroll
        for (int i = 0; i < 8; ++i) lstat[i] = lstat[i] * alpha[i] + rs[i];

        // restage P as bf16 (this wave's private 16x64 slice)
#pragma unroll
        for (int nt = 0; nt < 4; ++nt)
#pragma unroll
            for (int i = 0; i < 8; ++i)
                Ps[w * 16 + mb + i][nt * 16 + n] = (__bf16)sacc[nt][i];
        asm volatile("s_wait_dscnt 0" ::: "memory");

        // rescale O then accumulate P @ V
#pragma unroll
        for (int nt = 0; nt < 4; ++nt)
#pragma unroll
            for (int i = 0; i < 8; ++i) oacc[nt][i] *= alpha[i];

        const v16bf pa0 = load_a_frag(&Ps[w * 16][0], 72, lane);
        const v16bf pa1 = load_a_frag(&Ps[w * 16][32], 72, lane);
#pragma unroll
        for (int nt = 0; nt < 4; ++nt) {
            const v16bf v0 = load_bn_frag(&Vt[nt * 16][0], 72, lane);
            const v16bf v1 = load_bn_frag(&Vt[nt * 16][32], 72, lane);
            oacc[nt] = wmma_bf16(pa0, v0, oacc[nt]);
            oacc[nt] = wmma_bf16(pa1, v1, oacc[nt]);
        }
    }

    // normalize and restage O for the output projection
#pragma unroll
    for (int nt = 0; nt < 4; ++nt)
#pragma unroll
        for (int i = 0; i < 8; ++i)
            Ps[w * 16 + mb + i][nt * 16 + n] = (__bf16)(oacc[nt][i] / lstat[i]);
    asm volatile("s_wait_dscnt 0" ::: "memory");

    const v16bf oa0 = load_a_frag(&Ps[w * 16][0], 72, lane);
    const v16bf oa1 = load_a_frag(&Ps[w * 16][32], 72, lane);
    v8f fin[4];
#pragma unroll
    for (int nt = 0; nt < 4; ++nt) {
        fin[nt] = vzero();
        const v16bf b0 = load_bn_frag(&Wos[nt * 16][0], 72, lane);
        const v16bf b1 = load_bn_frag(&Wos[nt * 16][32], 72, lane);
        fin[nt] = wmma_bf16(oa0, b0, fin[nt]);
        fin[nt] = wmma_bf16(oa1, b1, fin[nt]);
    }

    const int bb = bh >> 4;           // / HEADS
    const int hh = bh & 15;
#pragma unroll
    for (int nt = 0; nt < 4; ++nt) {
        const float bov = bo[nt * 16 + n];
#pragma unroll
        for (int i = 0; i < 8; ++i) {
            const int s = q0 + w * 16 + mb + i;
            out[((size_t)bb * SEQ + s) * DIM + hh * DH + nt * 16 + n] = fin[nt][i] + bov;
        }
    }
}

// ---------------------------------------------------------------------------
extern "C" void kernel_launch(void* const* d_in, const int* in_sizes, int n_in,
                              void* d_out, int out_size, void* d_ws, size_t ws_size,
                              hipStream_t stream) {
    const float* q   = (const float*)d_in[0];
    const float* k   = (const float*)d_in[1];
    const float* v   = (const float*)d_in[2];
    const float* Wq  = (const float*)d_in[3];
    const float* bq  = (const float*)d_in[4];
    const float* Wk  = (const float*)d_in[5];
    const float* bk  = (const float*)d_in[6];
    const float* Wv  = (const float*)d_in[7];
    const float* bv  = (const float*)d_in[8];
    const float* Wqf = (const float*)d_in[9];
    const float* bqf = (const float*)d_in[10];
    const float* Wkf = (const float*)d_in[11];
    const float* bkf = (const float*)d_in[12];
    const float* Wvf = (const float*)d_in[13];
    const float* bvf = (const float*)d_in[14];
    const float* Wo  = (const float*)d_in[15];
    const float* bo  = (const float*)d_in[16];
    const int*   msk = (const int*)d_in[17];

    const size_t headElems = (size_t)BH * SEQ * DH;   // 8,388,608
    __bf16* Qh = (__bf16*)d_ws;
    __bf16* Kh = Qh + headElems;
    __bf16* Vh = Kh + headElems;

    dim3 blk(256);
    dim3 g1(HEADS, (BATCH * SEQ) / 128);
    // fold 1/sqrt(DH)=0.125 into Q (commutes with ReLU: positive scale)
    proj_ff_kernel<<<g1, blk, 0, stream>>>(q, Wq, bq, Wqf, bqf, Qh, 0.125f);
    proj_ff_kernel<<<g1, blk, 0, stream>>>(k, Wk, bk, Wkf, bkf, Kh, 1.0f);
    proj_ff_kernel<<<g1, blk, 0, stream>>>(v, Wv, bv, Wvf, bvf, Vh, 1.0f);

    dim3 g2(SEQ / 128, BH);
    attn_kernel<<<g2, blk, 0, stream>>>(Qh, Kh, Vh, msk, Wo, bo, (float*)d_out);
}
